// DHGFormer_28613072126236
// MI455X (gfx1250) — matrix operand order repaired and checked by reading, not compiled, not run on hardware
//
#include <hip/hip_runtime.h>
#include <hip/hip_bf16.h>
#include <stdint.h>

typedef __bf16 v16bf __attribute__((ext_vector_type(16)));
typedef __bf16 v8bf  __attribute__((ext_vector_type(8)));
typedef float  v8f   __attribute__((ext_vector_type(8)));

#define ROI 200
#define ROIP 208            // padded to 13*16
#define BATCH 1024
#define EMB 128

__constant__ int   c_start[8]  = {0, 41, 70, 91, 110, 130, 137, 158};
__constant__ int   c_end[8]    = {41, 70, 91, 110, 130, 137, 158, 200};
__constant__ float c_invlen[8] = {1.f/41.f, 1.f/29.f, 1.f/21.f, 1.f/19.f,
                                  1.f/20.f, 1.f/7.f,  1.f/21.f, 1.f/42.f};

__device__ __forceinline__ int segof(int i) {
    if (i < 41)  return 0;
    if (i < 70)  return 1;
    if (i < 91)  return 2;
    if (i < 110) return 3;
    if (i < 130) return 4;
    if (i < 137) return 5;
    if (i < 158) return 6;
    return 7;
}

// f32 -> bf16 round-to-nearest-even
__device__ __forceinline__ __bf16 f2bf(float f) {
    union { float f; uint32_t u; } c; c.f = f;
    uint32_t u = c.u + 0x7fffu + ((c.u >> 16) & 1u);
    uint16_t h = (uint16_t)(u >> 16);
    return __builtin_bit_cast(__bf16, h);
}
__device__ __forceinline__ float b2f(__bf16 h) {
    uint16_t u = __builtin_bit_cast(uint16_t, h);
    union { uint32_t u; float f; } c; c.u = ((uint32_t)u) << 16;
    return c.f;
}
__device__ __forceinline__ v8f wmma_bf16(v16bf a, v16bf b, v8f c) {
    return __builtin_amdgcn_wmma_f32_16x16x32_bf16(false, a, false, b, (short)0, c,
                                                   false, false);
}
__device__ __forceinline__ v16bf cat8(v8bf lo, v8bf hi) {
    return __builtin_shufflevector(lo, hi, 0, 1, 2, 3, 4, 5, 6, 7,
                                   8, 9, 10, 11, 12, 13, 14, 15);
}

// ---------------------------------------------------------------------------
// 1) softmax over E=128; emit zero-padded (208 rows/batch) bf16 hi/lo planes
//    one wave per padded row, 4 elements per lane
// ---------------------------------------------------------------------------
__global__ void softmax_bf16_kernel(const float* __restrict__ emb,
                                    __bf16* __restrict__ Phi,
                                    __bf16* __restrict__ Plo) {
    int row = blockIdx.x * blockDim.y + threadIdx.y;       // 0 .. 1024*208-1
    if (row >= BATCH * ROIP) return;
    int b = row / ROIP, roi = row - b * ROIP;
    int lane = threadIdx.x;
    __bf16* po = Phi + (long long)row * EMB + lane * 4;
    __bf16* qo = Plo + (long long)row * EMB + lane * 4;
    if (roi >= ROI) {                                       // wave-uniform pad path
        __bf16 z = (__bf16)0.0f;
#pragma unroll
        for (int j = 0; j < 4; ++j) { po[j] = z; qo[j] = z; }
        return;
    }
    const float* x = emb + ((long long)b * ROI + roi) * EMB + lane * 4;
    float v0 = x[0], v1 = x[1], v2 = x[2], v3 = x[3];
    float m = fmaxf(fmaxf(v0, v1), fmaxf(v2, v3));
    for (int off = 16; off >= 1; off >>= 1) m = fmaxf(m, __shfl_xor(m, off, 32));
    float e0 = __expf(v0 - m), e1 = __expf(v1 - m);
    float e2 = __expf(v2 - m), e3 = __expf(v3 - m);
    float s = e0 + e1 + e2 + e3;
    for (int off = 16; off >= 1; off >>= 1) s += __shfl_xor(s, off, 32);
    float inv = 1.f / s;
    float p[4] = {e0 * inv, e1 * inv, e2 * inv, e3 * inv};
#pragma unroll
    for (int j = 0; j < 4; ++j) {
        __bf16 hi = f2bf(p[j]);
        po[j] = hi;
        qo[j] = f2bf(p[j] - b2f(hi));
    }
}

// ---------------------------------------------------------------------------
// 2) A[b] = P[b] P[b]^T with zero-padded P: NO guards, direct 16B vector
//    global loads in the exact WMMA fragment layout; fully unrolled K.
// ---------------------------------------------------------------------------
__global__ void gemm_aat_kernel(const __bf16* __restrict__ Phi,
                                const __bf16* __restrict__ Plo,
                                float* __restrict__ A) {
    int b  = blockIdx.z;
    int tm = blockIdx.x;
    int tn = blockIdx.y * 8 + threadIdx.y;
    if (tn >= 13) return;                                   // wave-uniform
    int lane = threadIdx.x, mloc = lane & 15, hi = lane >> 4;
    long long base = (long long)b * ROIP * EMB;
    const __bf16* ph = Phi + base;
    const __bf16* pl = Plo + base;
    long long arow = (long long)(tm * 16 + mloc) * EMB;     // A-frag source row
    long long brow = (long long)(tn * 16 + mloc) * EMB;     // B-frag source row
    v8f c = {0.f, 0.f, 0.f, 0.f, 0.f, 0.f, 0.f, 0.f};
#pragma unroll
    for (int k0 = 0; k0 < EMB; k0 += 32) {
        const v8bf* pah = (const v8bf*)(ph + arow + k0 + hi * 8);
        const v8bf* pal = (const v8bf*)(pl + arow + k0 + hi * 8);
        const v8bf* pbh = (const v8bf*)(ph + brow + k0 + hi * 8);
        const v8bf* pbl = (const v8bf*)(pl + brow + k0 + hi * 8);
        v16bf ah = cat8(pah[0], pah[2]);   // k0+hi*8+0..7 , k0+16+hi*8+0..7
        v16bf al = cat8(pal[0], pal[2]);
        v16bf bh = cat8(pbh[0], pbh[2]);   // k0+hi*8+0..7 , k0+16+hi*8+0..7
        v16bf bl = cat8(pbl[0], pbl[2]);
        c = wmma_bf16(ah, bh, c);
        c = wmma_bf16(ah, bl, c);
        c = wmma_bf16(al, bh, c);
    }
    float* Ab = A + (long long)b * ROI * ROI;
    int nout = tn * 16 + mloc;
#pragma unroll
    for (int r = 0; r < 8; ++r) {
        int mout = tm * 16 + r + (hi << 3);
        if (mout < ROI && nout < ROI) Ab[(long long)mout * ROI + nout] = c[r];
    }
}

// ---------------------------------------------------------------------------
// 3) generic batched Y[b] = act(X[b](MxK) * W^T(F,K) + bias)
//    LDS-staged split-bf16; clamped unconditional loads + register selects.
//    8 waves/block: shared X tile (tm), per-wave W tile (tn).
// ---------------------------------------------------------------------------
#define SSTR 40   // LDS row stride in halves: 80B (16B aligned, conflict-free)

__global__ void gemm_xw_kernel(const float* __restrict__ X,
                               const float* __restrict__ W,
                               const float* __restrict__ bias,
                               float* __restrict__ Y,
                               int M, int K, int F, int act) {
    __shared__ __attribute__((aligned(16))) __bf16 sAh[16 * SSTR];
    __shared__ __attribute__((aligned(16))) __bf16 sAl[16 * SSTR];
    __shared__ __attribute__((aligned(16))) __bf16 sBh[8 * 16 * SSTR];
    __shared__ __attribute__((aligned(16))) __bf16 sBl[8 * 16 * SSTR];

    int tm = blockIdx.x;
    int b  = blockIdx.z;
    int w  = threadIdx.y;
    int tn = blockIdx.y * 8 + w;
    int Ftiles = (F + 15) >> 4;
    bool active = (tn < Ftiles);
    int lane = threadIdx.x, mloc = lane & 15, hi = lane >> 4;
    int tid = w * 32 + lane;
    long long xbase = (long long)b * M * K;
    long long ybase = (long long)b * M * F;
    v8f c = {0.f, 0.f, 0.f, 0.f, 0.f, 0.f, 0.f, 0.f};

    for (int k0 = 0; k0 < K; k0 += 32) {
        // ---- stage X tile: 16 rows x 32 k (2 elems/thread) ----
#pragma unroll
        for (int it = 0; it < 2; ++it) {
            int idx = tid + it * 256;            // 0..511
            int m = idx >> 5, kloc = idx & 31;
            int mg = tm * 16 + m, k = k0 + kloc;
            float v = X[xbase + (long long)min(mg, M - 1) * K + min(k, K - 1)];
            v = (mg < M && k < K) ? v : 0.f;     // register select, no branch
            __bf16 h = f2bf(v);
            sAh[m * SSTR + kloc] = h;
            sAl[m * SSTR + kloc] = f2bf(v - b2f(h));
        }
        // ---- stage 8 W tiles: 8 x 16 rows x 32 k (16 elems/thread) ----
        int myF0 = blockIdx.y * 128;             // first f of this block
#pragma unroll
        for (int it = 0; it < 16; ++it) {
            int idx = tid + it * 256;            // 0..4095
            int ww = idx >> 9, rem = idx & 511;
            int n = rem >> 5, kloc = rem & 31;
            int fg = myF0 + ww * 16 + n, k = k0 + kloc;
            float v = W[(long long)min(fg, F - 1) * K + min(k, K - 1)];
            v = (fg < F && k < K) ? v : 0.f;
            __bf16 h = f2bf(v);
            sBh[(ww * 16 + n) * SSTR + kloc] = h;
            sBl[(ww * 16 + n) * SSTR + kloc] = f2bf(v - b2f(h));
        }
        __syncthreads();
        if (active) {                            // wave-uniform; EXEC all ones
            const v8bf* ah0 = (const v8bf*)&sAh[mloc * SSTR + hi * 8];
            const v8bf* ah1 = (const v8bf*)&sAh[mloc * SSTR + 16 + hi * 8];
            const v8bf* al0 = (const v8bf*)&sAl[mloc * SSTR + hi * 8];
            const v8bf* al1 = (const v8bf*)&sAl[mloc * SSTR + 16 + hi * 8];
            const v8bf* bh0 = (const v8bf*)&sBh[(w * 16 + mloc) * SSTR + hi * 8];
            const v8bf* bl0 = (const v8bf*)&sBl[(w * 16 + mloc) * SSTR + hi * 8];
            v16bf ah = cat8(ah0[0], ah1[0]);
            v16bf al = cat8(al0[0], al1[0]);
            v16bf bh = cat8(bh0[0], bh0[2]);
            v16bf bl = cat8(bl0[0], bl0[2]);
            c = wmma_bf16(ah, bh, c);
            c = wmma_bf16(ah, bl, c);
            c = wmma_bf16(al, bh, c);
        }
        __syncthreads();
    }
    if (active) {
        int nout = tn * 16 + mloc;
#pragma unroll
        for (int r = 0; r < 8; ++r) {
            int mout = tm * 16 + r + (hi << 3);
            if (mout < M && nout < F) {
                float v = c[r] + bias[nout];
                if (act) v = (v >= 0.f) ? v : 0.2f * v;
                Y[ybase + (long long)mout * F + nout] = v;
            }
        }
    }
}

// ---------------------------------------------------------------------------
// 4) prop_prep: rs_intra[b,i], rs_inter[b,s] from A
// ---------------------------------------------------------------------------
__global__ void prop_prep_kernel(const float* __restrict__ A,
                                 float* __restrict__ rsA, float* __restrict__ rsI) {
    int b = blockIdx.x;
    const float* Ab = A + (long long)b * ROI * ROI;
    __shared__ float rw[ROI];
    int i = threadIdx.x;
    if (i < ROI) {
        int s = segof(i);
        int s0 = c_start[s], s1 = c_end[s];
        float intra = 0.f, wsum = 0.f;
        for (int j = 0; j < ROI; ++j) {
            float a = Ab[i * ROI + j];
            wsum += a * c_invlen[segof(j)];
            if (j >= s0 && j < s1) intra += a;
        }
        rsA[b * ROI + i] = intra;
        rw[i] = wsum;
    }
    __syncthreads();
    if (i < 8) {
        float s = 0.f;
        for (int k = c_start[i]; k < c_end[i]; ++k) s += rw[k];
        rsI[b * 8 + i] = s * c_invlen[i];
    }
}

// ---------------------------------------------------------------------------
// 5) prop: out = 0.5*(rs_intra*x + rs_inter[seg]*blockmean(x)[seg]), F=200
// ---------------------------------------------------------------------------
__global__ void prop_kernel(const float* __restrict__ xin,
                            const float* __restrict__ rsA,
                            const float* __restrict__ rsI,
                            float* __restrict__ xout) {
    const int F = ROI;
    int b = blockIdx.x;
    const float* x = xin + (long long)b * ROI * F;
    float* y = xout + (long long)b * ROI * F;
    __shared__ float ssum[8 * ROI];
    for (int t = threadIdx.x; t < 8 * ROI; t += blockDim.x) ssum[t] = 0.f;
    __syncthreads();
    for (int idx = threadIdx.x; idx < ROI * F; idx += blockDim.x) {
        int i = idx / F, f = idx - i * F;
        atomicAdd(&ssum[segof(i) * ROI + f], x[idx]);
    }
    __syncthreads();
    for (int idx = threadIdx.x; idx < ROI * F; idx += blockDim.x) {
        int i = idx / F, f = idx - i * F;
        int s = segof(i);
        float sub = rsI[b * 8 + s] * ssum[s * ROI + f] * c_invlen[s];
        y[idx] = 0.5f * (rsA[b * ROI + i] * x[idx] + sub);
    }
}

// ---------------------------------------------------------------------------
// 6) BatchNorm over rows (features last dim = 200): deterministic two-phase
// ---------------------------------------------------------------------------
__global__ void bn_partial_kernel(const float* __restrict__ x,
                                  float* __restrict__ part,
                                  int rowsPerBlock, int nrows) {
    const int F = ROI;
    int t = threadIdx.x;
    float s = 0.f, ss = 0.f;
    if (t < F) {
        int r0 = blockIdx.x * rowsPerBlock;
        int r1 = min(r0 + rowsPerBlock, nrows);
        for (int r = r0; r < r1; ++r) {
            float v = x[(long long)r * F + t];
            s += v; ss += v * v;
        }
    }
    part[(long long)blockIdx.x * 512 + t]       = s;
    part[(long long)blockIdx.x * 512 + 256 + t] = ss;
}

__global__ void bn_finalize_kernel(const float* __restrict__ part, int G, float nrows,
                                   float* __restrict__ bnm, float* __restrict__ bnr) {
    int t = threadIdx.x;
    if (t >= ROI) return;
    float s = 0.f, ss = 0.f;
    for (int g = 0; g < G; ++g) {
        s  += part[(long long)g * 512 + t];
        ss += part[(long long)g * 512 + 256 + t];
    }
    float m = s / nrows;
    float var = ss / nrows - m * m;
    bnm[t] = m;
    bnr[t] = rsqrtf(var + 1e-5f);
}

__global__ void bn_apply_kernel(float* __restrict__ x, long long total,
                                const float* __restrict__ bnm,
                                const float* __restrict__ bnr,
                                const float* __restrict__ g,
                                const float* __restrict__ bvec) {
    const int F = ROI;
    for (long long idx = (long long)blockIdx.x * blockDim.x + threadIdx.x; idx < total;
         idx += (long long)gridDim.x * blockDim.x) {
        int f = (int)(idx % F);
        x[idx] = (x[idx] - bnm[f]) * bnr[f] * g[f] + bvec[f];
    }
}

// ---------------------------------------------------------------------------
// 7) bn3: per-node stats over (B, 8); deterministic shared tree
// ---------------------------------------------------------------------------
__global__ void bn3_stats_kernel(const float* __restrict__ x,
                                 float* __restrict__ mean, float* __restrict__ rstd) {
    int n = blockIdx.x, t = threadIdx.x;
    float s = 0.f, ss = 0.f;
    for (int b = t; b < BATCH; b += 256) {
        const float* p = x + (long long)b * (ROI * 8) + n * 8;
#pragma unroll
        for (int f = 0; f < 8; ++f) { float v = p[f]; s += v; ss += v * v; }
    }
    __shared__ float r1[256], r2[256];
    r1[t] = s; r2[t] = ss;
    __syncthreads();
    for (int o = 128; o >= 1; o >>= 1) {
        if (t < o) { r1[t] += r1[t + o]; r2[t] += r2[t + o]; }
        __syncthreads();
    }
    if (t == 0) {
        float N = (float)(BATCH * 8);
        float m = r1[0] / N;
        float var = r2[0] / N - m * m;
        mean[n] = m;
        rstd[n] = rsqrtf(var + 1e-5f);
    }
}

__global__ void bn3_apply_kernel(float* __restrict__ x, long long total,
                                 const float* __restrict__ mean,
                                 const float* __restrict__ rstd,
                                 const float* __restrict__ g,
                                 const float* __restrict__ bvec) {
    for (long long idx = (long long)blockIdx.x * blockDim.x + threadIdx.x; idx < total;
         idx += (long long)gridDim.x * blockDim.x) {
        int n = (int)((idx % (ROI * 8)) >> 3);
        x[idx] = (x[idx] - mean[n]) * rstd[n] * g[n] + bvec[n];
    }
}

// ---------------------------------------------------------------------------
// 8) edge variance: per-batch var (ddof=1) then fixed-order mean
// ---------------------------------------------------------------------------
__global__ void edgevar_kernel(const float* __restrict__ A, float* __restrict__ bvar) {
    int b = blockIdx.x, t = threadIdx.x;
    const float* Ab = A + (long long)b * ROI * ROI;
    float s = 0.f, ss = 0.f;
    for (int i = t; i < ROI * ROI; i += 256) {
        float v = Ab[i];
        s += v; ss += v * v;
    }
    __shared__ float r1[256], r2[256];
    r1[t] = s; r2[t] = ss;
    __syncthreads();
    for (int o = 128; o >= 1; o >>= 1) {
        if (t < o) { r1[t] += r1[t + o]; r2[t] += r2[t + o]; }
        __syncthreads();
    }
    if (t == 0) {
        float n = (float)(ROI * ROI);
        float m = r1[0] / n;
        bvar[b] = (r2[0] - n * m * m) / (n - 1.f);
    }
}

__global__ void edgevar_final_kernel(const float* __restrict__ bvar, float* __restrict__ ev) {
    int t = threadIdx.x;
    float s = 0.f;
    for (int i = t; i < BATCH; i += 256) s += bvar[i];
    __shared__ float r1[256];
    r1[t] = s;
    __syncthreads();
    for (int o = 128; o >= 1; o >>= 1) {
        if (t < o) r1[t] += r1[t + o];
        __syncthreads();
    }
    if (t == 0) ev[0] = r1[0] / (float)BATCH;
}

// ---------------------------------------------------------------------------
extern "C" void kernel_launch(void* const* d_in, const int* in_sizes, int n_in,
                              void* d_out, int out_size, void* d_ws, size_t ws_size,
                              hipStream_t stream) {
    (void)in_sizes; (void)n_in; (void)out_size; (void)ws_size;
    const float* emb   = (const float*)d_in[0];
    const float* nf    = (const float*)d_in[1];
    const float* W_g0  = (const float*)d_in[2];
    const float* b_g0  = (const float*)d_in[3];
    const float* W_g1  = (const float*)d_in[4];
    const float* b_g1  = (const float*)d_in[5];
    const float* bn1_g = (const float*)d_in[6];
    const float* bn1_b = (const float*)d_in[7];
    const float* W_gc1 = (const float*)d_in[8];
    const float* b_gc1 = (const float*)d_in[9];
    const float* bn2_g = (const float*)d_in[10];
    const float* bn2_b = (const float*)d_in[11];
    const float* W_2a  = (const float*)d_in[12];
    const float* b_2a  = (const float*)d_in[13];
    const float* W_2b  = (const float*)d_in[14];
    const float* b_2b  = (const float*)d_in[15];
    const float* bn3_g = (const float*)d_in[16];
    const float* bn3_b = (const float*)d_in[17];
    const float* W_c1  = (const float*)d_in[18];
    const float* b_c1  = (const float*)d_in[19];
    const float* W_c2  = (const float*)d_in[20];
    const float* b_c2  = (const float*)d_in[21];
    const float* W_c3  = (const float*)d_in[22];
    const float* b_c3  = (const float*)d_in[23];

    float* pred = (float*)d_out;                 // [1024,2]
    float* A    = pred + 2048;                   // [1024,200,200]
    float* ev   = pred + 2048 + (long long)BATCH * ROI * ROI;  // scalar

    // workspace layout (floats)
    const long long XB = (long long)BATCH * ROI * ROI;         // 40,960,000
    float* ws   = (float*)d_ws;
    float* x0   = ws;
    float* x1   = ws + XB;
    float* tail = ws + 2 * XB;
    float* rsA  = tail;                    // 1024*200
    float* rsI  = rsA + BATCH * ROI;       // 1024*8
    float* part = rsI + BATCH * 8;         // 400*512
    float* bnm  = part + 400 * 512;        // 256
    float* bnr  = bnm + 256;               // 256
    float* bvar = bnr + 256;               // 1024
    float* y1   = bvar + BATCH;            // 1024*256
    float* y2   = y1 + BATCH * 256;        // 1024*32
    __bf16* Phi = (__bf16*)x1;             // padded planes alias x1 (dead before x1 written)
    __bf16* Plo = Phi + (long long)BATCH * ROIP * EMB;

    dim3 wblk(32, 8);
    const int nrows = BATCH * ROI;         // 204800

    // 1) softmax -> zero-padded bf16 hi/lo planes (1024*208 rows)
    softmax_bf16_kernel<<<(BATCH * ROIP) / 8, wblk, 0, stream>>>(emb, Phi, Plo);
    // 2) A = P P^T  (13x13 tiles, 8 waves/block, guard-free vector loads)
    gemm_aat_kernel<<<dim3(13, 2, BATCH), wblk, 0, stream>>>(Phi, Plo, A);
    // 3) propagation scalars
    prop_prep_kernel<<<BATCH, 256, 0, stream>>>(A, rsA, rsI);
    // 4) x = prop(node_features)
    prop_kernel<<<BATCH, 256, 0, stream>>>(nf, rsA, rsI, x0);
    // 5) x = lrelu(x @ Wg0^T + b) ; then @ Wg1^T + b (no act)
    gemm_xw_kernel<<<dim3(13, 2, BATCH), wblk, 0, stream>>>(x0, W_g0, b_g0, x1, ROI, ROI, ROI, 1);
    gemm_xw_kernel<<<dim3(13, 2, BATCH), wblk, 0, stream>>>(x1, W_g1, b_g1, x0, ROI, ROI, ROI, 0);
    // 6) bn1 (in-place on x0)
    bn_partial_kernel<<<400, 256, 0, stream>>>(x0, part, 512, nrows);
    bn_finalize_kernel<<<1, 256, 0, stream>>>(part, 400, (float)nrows, bnm, bnr);
    bn_apply_kernel<<<4096, 256, 0, stream>>>(x0, XB, bnm, bnr, bn1_g, bn1_b);
    // 7) prop, gc1 (lrelu), bn2
    prop_kernel<<<BATCH, 256, 0, stream>>>(x0, rsA, rsI, x1);
    gemm_xw_kernel<<<dim3(13, 2, BATCH), wblk, 0, stream>>>(x1, W_gc1, b_gc1, x0, ROI, ROI, ROI, 1);
    bn_partial_kernel<<<400, 256, 0, stream>>>(x0, part, 512, nrows);
    bn_finalize_kernel<<<1, 256, 0, stream>>>(part, 400, (float)nrows, bnm, bnr);
    bn_apply_kernel<<<4096, 256, 0, stream>>>(x0, XB, bnm, bnr, bn2_g, bn2_b);
    // 8) prop, gcn2: 200->64 (lrelu), 64->8 (lrelu)
    prop_kernel<<<BATCH, 256, 0, stream>>>(x0, rsA, rsI, x1);
    gemm_xw_kernel<<<dim3(13, 1, BATCH), wblk, 0, stream>>>(x1, W_2a, b_2a, x0, ROI, ROI, 64, 1);
    gemm_xw_kernel<<<dim3(13, 1, BATCH), wblk, 0, stream>>>(x0, W_2b, b_2b, x1, ROI, 64, 8, 1);
    // 9) bn3 (per-node stats over B x 8), in-place on x1
    bn3_stats_kernel<<<ROI, 256, 0, stream>>>(x1, bnm, bnr);
    bn3_apply_kernel<<<6400, 256, 0, stream>>>(x1, (long long)BATCH * ROI * 8, bnm, bnr, bn3_g, bn3_b);
    // 10) classifier: [1024,1600] -> 256 -> 32 -> 2
    gemm_xw_kernel<<<dim3(64, 2, 1), wblk, 0, stream>>>(x1, W_c1, b_c1, y1, BATCH, ROI * 8, 256, 1);
    gemm_xw_kernel<<<dim3(64, 1, 1), wblk, 0, stream>>>(y1, W_c2, b_c2, y2, BATCH, 256, 32, 1);
    gemm_xw_kernel<<<dim3(64, 1, 1), wblk, 0, stream>>>(y2, W_c3, b_c3, pred, BATCH, 32, 2, 0);
    // 11) edge variance
    edgevar_kernel<<<BATCH, 256, 0, stream>>>(A, bvar);
    edgevar_final_kernel<<<1, 256, 0, stream>>>(bvar, ev);
}
